// GraphAttentionLayer_43293270344083
// MI455X (gfx1250) — compile-verified, matrix operand order checked by reading, not run on hardware
//
#include <hip/hip_runtime.h>
#include <hip/hip_bf16.h>
#include <stdint.h>

#define N_NODES 8192
#define FIN     512
#define FOUT    64
#define ALPHA   0.2f
#define NEG_INF -9.0e15f

typedef __attribute__((ext_vector_type(16))) __bf16 v16bf;
typedef __attribute__((ext_vector_type(8)))  float  v8f;
typedef __attribute__((ext_vector_type(2)))  float  v2f;
typedef __attribute__((ext_vector_type(4)))  int    vint4;

// K-pair base index for 16-bit A/B WMMA fragment slot v (0..7), lane half hf.
// ISA 7.12.2 (16-bit 16x32 A): lanes 0-15: V0..3 -> K={0..7}, V4..7 -> K={16..23};
// lanes 16-31: V0..3 -> K={8..15}, V4..7 -> K={24..31}.  (Mirror layout for B.)
__device__ __forceinline__ int kbase(int v, int hf) {
    return (v < 4 ? 2 * v : 16 + 2 * (v - 4)) + 8 * hf;
}

// round-to-nearest-even f32 -> bf16, packed pair
__device__ __forceinline__ uint32_t f2bf_pack(float lo, float hi) {
    union { float f; uint32_t u; } a, b;
    a.f = lo; b.f = hi;
    uint32_t ulo = (a.u + 0x7FFFu + ((a.u >> 16) & 1u)) >> 16;
    uint32_t uhi = (b.u + 0x7FFFu + ((b.u >> 16) & 1u)) >> 16;
    return (uhi << 16) | (ulo & 0xFFFFu);
}

// ---------------------------------------------------------------------------
// Kernel 1: h = input @ W via V_WMMA_F32_16X16X4_F32.
// One wave per 16x16 output tile; K loop in steps of 4.
// 32-bit A (16x4): lanes 0-15 hold K=0,1 in V0,V1; lanes 16-31 hold K=2,3.
// 32-bit B (4x16): V0 = row K=2*hf, V1 = row K=2*hf+1, col = lane&15.
// ---------------------------------------------------------------------------
__global__ __launch_bounds__(32) void h_gemm_wmma_kernel(
    const float* __restrict__ input, const float* __restrict__ W,
    float* __restrict__ h)
{
    const int lane = threadIdx.x & 31;
    const int rb = blockIdx.x >> 2;      // row block (512)
    const int cb = blockIdx.x & 3;       // col block (4)
    const int row0 = rb * 16, col0 = cb * 16;
    const int m  = lane & 15;
    const int hf = lane >> 4;

    v8f acc = {};
    const float* arow = input + (size_t)(row0 + m) * FIN + 2 * hf;
    #pragma unroll 4
    for (int kb = 0; kb < FIN; kb += 4) {
        v2f a, b;
        a.x = arow[kb];
        a.y = arow[kb + 1];
        b.x = W[(size_t)(kb + 2 * hf    ) * FOUT + col0 + m];
        b.y = W[(size_t)(kb + 2 * hf + 1) * FOUT + col0 + m];
        acc = __builtin_amdgcn_wmma_f32_16x16x4_f32(false, a, false, b,
                                                    (short)0, acc, false, false);
    }
    // C/D layout: VGPR r -> M = r + 8*hf, N = lane&15
    #pragma unroll
    for (int r = 0; r < 8; ++r)
        h[(size_t)(row0 + r + 8 * hf) * FOUT + col0 + m] = acc[r];
}

// ---------------------------------------------------------------------------
// Kernel 2: s[i] = h[i,:].a_src, d[i] = h[i,:].a_dst
// ---------------------------------------------------------------------------
__global__ void sd_kernel(const float* __restrict__ h, const float* __restrict__ a,
                          float* __restrict__ s, float* __restrict__ d)
{
    int i = blockIdx.x * blockDim.x + threadIdx.x;
    if (i >= N_NODES) return;
    float ss = 0.f, dd = 0.f;
    #pragma unroll
    for (int k = 0; k < FOUT; ++k) {
        float hv = h[(size_t)i * FOUT + k];
        ss = fmaf(hv, a[k], ss);
        dd = fmaf(hv, a[FOUT + k], dd);
    }
    s[i] = ss; d[i] = dd;
}

// ---------------------------------------------------------------------------
// Kernel 3: pre-shuffle h into the bf16 A-fragment layout for the attention
// WMMA (A = h^T tile: M = feature-in-group, K = j).
// hA[((jtile*4+g)*32+lane)*8 + v] = pack(h[jbase+k][n], h[jbase+k+1][n])
// with n = g*16 + (lane&15), k = kbase(v, lane>>4).
// ---------------------------------------------------------------------------
__global__ void pack_hA_kernel(const float* __restrict__ h, uint32_t* __restrict__ hA)
{
    int t = blockIdx.x * blockDim.x + threadIdx.x;   // (jtile*4+g)*32 + lane
    int lane = t & 31;
    int g  = (t >> 5) & 3;
    int jt = t >> 7;
    if (jt >= N_NODES / 32) return;
    int n  = g * 16 + (lane & 15);
    int hf = lane >> 4;
    int jbase = jt * 32;
    uint32_t* dst = hA + (size_t)t * 8;
    #pragma unroll
    for (int v = 0; v < 8; ++v) {
        int k = kbase(v, hf);
        dst[v] = f2bf_pack(h[(size_t)(jbase + k) * FOUT + n],
                           h[(size_t)(jbase + k + 1) * FOUT + n]);
    }
}

// ---------------------------------------------------------------------------
// Kernel 4: fused masked-softmax attention, flash-attention style, TRANSPOSED:
// D = hA (16 feat x 32 j) x P^T (32 j x 16 rows), so the C fragment has
// N = attention row = lane&15 — per-row softmax rescale needs no cross-lane
// traffic.  Block = 256 threads (8 waves) per 16 output rows; each wave does
// online softmax over its 1024-column span; partials merged through LDS.
// adj is streamed with non-temporal loads (keeps hA/d resident in L2).
// ---------------------------------------------------------------------------
__global__ __launch_bounds__(256) void gat_attn_kernel(
    const int* __restrict__ adj, const float* __restrict__ s,
    const float* __restrict__ d, const uint32_t* __restrict__ hA,
    float* __restrict__ out)
{
    __shared__ float acc_sh[8][16][FOUT];
    __shared__ float m_sh[8][16];
    __shared__ float l_sh[8][16];

    const int lane = threadIdx.x & 31;
    const int wave = threadIdx.x >> 5;
    const int row0 = blockIdx.x * 16;
    const int m    = lane & 15;       // attention row handled by this lane
    const int hf   = lane >> 4;       // lane half selects k-subranges

    const float s_m = s[row0 + m];

    v8f acc0 = {}, acc1 = {}, acc2 = {}, acc3 = {};
    float mrow = NEG_INF;
    float lrow = 0.f;

    const int jspan = N_NODES / 8;    // 1024 columns per wave
    const int j0 = wave * jspan;
    const int* adj_row = adj + (size_t)(row0 + m) * N_NODES;

    for (int jb = j0; jb < j0 + jspan; jb += 32) {
        const int off0 = jb + (hf ? 8 : 0);
        const int off1 = jb + (hf ? 24 : 16);

        // stream next adj chunk (global_prefetch_b8)
        __builtin_prefetch(adj_row + jb + 32, 0, 1);

        // non-temporal streaming loads of the adjacency mask
        vint4 a0 = __builtin_nontemporal_load((const vint4*)(adj_row + off0));
        vint4 a1 = __builtin_nontemporal_load((const vint4*)(adj_row + off0 + 4));
        vint4 a2 = __builtin_nontemporal_load((const vint4*)(adj_row + off1));
        vint4 a3 = __builtin_nontemporal_load((const vint4*)(adj_row + off1 + 4));
        float4 d0 = *(const float4*)(d + off0);
        float4 d1 = *(const float4*)(d + off0 + 4);
        float4 d2 = *(const float4*)(d + off1);
        float4 d3 = *(const float4*)(d + off1 + 4);

        int av[16]   = {a0.x,a0.y,a0.z,a0.w, a1.x,a1.y,a1.z,a1.w,
                        a2.x,a2.y,a2.z,a2.w, a3.x,a3.y,a3.z,a3.w};
        float dv[16] = {d0.x,d0.y,d0.z,d0.w, d1.x,d1.y,d1.z,d1.w,
                        d2.x,d2.y,d2.z,d2.w, d3.x,d3.y,d3.z,d3.w};

        // leaky-relu logits + masked tile max
        float ev[16];
        float tmax = NEG_INF;
        #pragma unroll
        for (int i = 0; i < 16; ++i) {
            float e = s_m + dv[i];
            e = e > 0.f ? e : ALPHA * e;
            ev[i] = e;
            if (av[i] > 0 && e > tmax) tmax = e;
        }
        // row tile-max across the two lane halves holding this row
        tmax = fmaxf(tmax, __shfl_xor(tmax, 16, 32));
        const float mnew  = fmaxf(mrow, tmax);
        const float scale = __expf(mrow - mnew);

        // probabilities -> bf16 B fragment (P^T, 32x16: N = lane&15 = row)
        union { uint32_t u[8]; v16bf v; } pfrag;
        float rsum = 0.f;
        #pragma unroll
        for (int v = 0; v < 8; ++v) {
            float p0 = av[2*v]     > 0 ? __expf(ev[2*v]     - mnew) : 0.f;
            float p1 = av[2*v + 1] > 0 ? __expf(ev[2*v + 1] - mnew) : 0.f;
            rsum += p0 + p1;
            pfrag.u[v] = f2bf_pack(p0, p1);
        }
        rsum += __shfl_xor(rsum, 16, 32);
        lrow = lrow * scale + rsum;
        mrow = mnew;

        // rescale accumulators: C fragment N = lane&15 = this lane's row,
        // so the scale is already resident in-lane.  No shuffles.
        #pragma unroll
        for (int r = 0; r < 8; ++r) {
            acc0[r] *= scale; acc1[r] *= scale; acc2[r] *= scale; acc3[r] *= scale;
        }

        // load 4 pre-shuffled h A-tiles (two b128 each) and WMMA
        const uint32_t* ha = hA + (size_t)(jb >> 5) * 4 * 32 * 8;
        union { uint4 q[2]; v16bf v; } h0, h1, h2, h3;
        h0.q[0] = ((const uint4*)(ha + (0 * 32 + lane) * 8))[0];
        h0.q[1] = ((const uint4*)(ha + (0 * 32 + lane) * 8))[1];
        h1.q[0] = ((const uint4*)(ha + (1 * 32 + lane) * 8))[0];
        h1.q[1] = ((const uint4*)(ha + (1 * 32 + lane) * 8))[1];
        h2.q[0] = ((const uint4*)(ha + (2 * 32 + lane) * 8))[0];
        h2.q[1] = ((const uint4*)(ha + (2 * 32 + lane) * 8))[1];
        h3.q[0] = ((const uint4*)(ha + (3 * 32 + lane) * 8))[0];
        h3.q[1] = ((const uint4*)(ha + (3 * 32 + lane) * 8))[1];

        acc0 = __builtin_amdgcn_wmma_f32_16x16x32_bf16(false, h0.v, false, pfrag.v,
                                                       (short)0, acc0, false, false);
        acc1 = __builtin_amdgcn_wmma_f32_16x16x32_bf16(false, h1.v, false, pfrag.v,
                                                       (short)0, acc1, false, false);
        acc2 = __builtin_amdgcn_wmma_f32_16x16x32_bf16(false, h2.v, false, pfrag.v,
                                                       (short)0, acc2, false, false);
        acc3 = __builtin_amdgcn_wmma_f32_16x16x32_bf16(false, h3.v, false, pfrag.v,
                                                       (short)0, acc3, false, false);
    }

    // spill wave-partials to LDS.  D fragment: VGPR r -> feature (g*16 + r + 8*hf),
    // N = lane&15 = attention row.
    #pragma unroll
    for (int r = 0; r < 8; ++r) {
        int f = r + 8 * hf;
        acc_sh[wave][m][ 0 + f] = acc0[r];
        acc_sh[wave][m][16 + f] = acc1[r];
        acc_sh[wave][m][32 + f] = acc2[r];
        acc_sh[wave][m][48 + f] = acc3[r];
    }
    if (lane < 16) { m_sh[wave][lane] = mrow; l_sh[wave][lane] = lrow; }
    __syncthreads();

    // combine the 8 wave-partials (flash-attention merge) and write out
    for (int idx = threadIdx.x; idx < 16 * FOUT; idx += 256) {
        int M  = idx >> 6;
        int nc = idx & 63;
        float mm = NEG_INF;
        #pragma unroll
        for (int w = 0; w < 8; ++w) mm = fmaxf(mm, m_sh[w][M]);
        float lsum = 0.f, vsum = 0.f;
        #pragma unroll
        for (int w = 0; w < 8; ++w) {
            float f = __expf(m_sh[w][M] - mm);
            lsum += l_sh[w][M] * f;
            vsum += acc_sh[w][M][nc] * f;
        }
        out[(size_t)(row0 + M) * FOUT + nc] = vsum / lsum;
    }
}

// ---------------------------------------------------------------------------
extern "C" void kernel_launch(void* const* d_in, const int* in_sizes, int n_in,
                              void* d_out, int out_size, void* d_ws, size_t ws_size,
                              hipStream_t stream)
{
    const float* input = (const float*)d_in[0];   // 8192 x 512 f32
    const int*   adj   = (const int*)d_in[1];     // 8192 x 8192 i32
    const float* W     = (const float*)d_in[2];   // 512 x 64 f32
    const float* a     = (const float*)d_in[3];   // 128 x 1 f32
    float* out = (float*)d_out;                   // 8192 x 64 f32

    char* ws = (char*)d_ws;
    float*    h  = (float*)ws;                               // 2 MB
    float*    s  = (float*)(ws + (size_t)N_NODES * FOUT * 4);
    float*    dd = s + N_NODES;
    uint32_t* hA = (uint32_t*)(dd + N_NODES);                // 1 MB

    h_gemm_wmma_kernel<<<(N_NODES / 16) * (FOUT / 16), 32, 0, stream>>>(input, W, h);
    sd_kernel<<<N_NODES / 256, 256, 0, stream>>>(h, a, s, dd);
    pack_hA_kernel<<<(N_NODES / 32) * 4 * 32 / 256, 256, 0, stream>>>(h, hA);
    gat_attn_kernel<<<N_NODES / 16, 256, 0, stream>>>(adj, s, dd, hA, out);
}
